// ConvPlus1d_78262894068116
// MI455X (gfx1250) — matrix operand order repaired and checked by reading, not compiled, not run on hardware
//
#include <hip/hip_runtime.h>
#include <stdint.h>

#define B_   32
#define CIN  64
#define COUT 128
#define KW   3
#define LEN  8192
#define KDIM (CIN * KW) /* 192 */

typedef __attribute__((ext_vector_type(16))) __bf16 v16bf;
typedef __attribute__((ext_vector_type(8)))  float  v8f;

union ABfrag {
    v16bf    v;
    uint32_t u[8];
    uint16_t h[16];
};

static __device__ __forceinline__ uint16_t f2bf(float f) {
    uint32_t u = __builtin_bit_cast(uint32_t, f);
    u += 0x7fffu + ((u >> 16) & 1u);   // round-to-nearest-even
    return (uint16_t)(u >> 16);
}

// ---------------------------------------------------------------------------
// Kernel 1: per-(b,c) row sum -> kern[b,c,j] via the analytic mean-of-conv
// kern[b,c,j] = (1/L) * (wk0*(S - x[L-1]) + wk1*S + wk2*(S - x[0]))
// ---------------------------------------------------------------------------
__global__ __launch_bounds__(256) void k_rowsum(const float* __restrict__ x,
                                                const float* __restrict__ Wk,
                                                float* __restrict__ kern) {
    __shared__ float red[256];
    const int b = blockIdx.y, c = blockIdx.x;
    const int tid = threadIdx.x;
    const float* row = x + ((size_t)b * CIN + c) * LEN;
    const float4* r4 = reinterpret_cast<const float4*>(row);
    float s = 0.f;
    for (int i = tid; i < LEN / 4; i += 256) {
        float4 v = r4[i];
        s += v.x + v.y + v.z + v.w;
    }
    red[tid] = s;
    __syncthreads();
    for (int off = 128; off > 0; off >>= 1) {
        if (tid < off) red[tid] += red[tid + off];
        __syncthreads();
    }
    if (tid == 0) {
        const float S = red[0];
        const float x0 = row[0], xl = row[LEN - 1];
        const float inv = 1.f / (float)LEN;
#pragma unroll
        for (int j = 0; j < KW; ++j) {
            const float* wk = Wk + (size_t)(c * KW + j) * KW;
            kern[((size_t)b * CIN + c) * KW + j] =
                (wk[0] * (S - xl) + wk[1] * S + wk[2] * (S - x0)) * inv;
        }
    }
}

// ---------------------------------------------------------------------------
// Kernel 2: synthesize per-sample conv weights (bf16 [COUT][KDIM]) and bias
// ---------------------------------------------------------------------------
__global__ __launch_bounds__(256) void k_makew(const float* __restrict__ kern,
                                               const float* __restrict__ W_in,
                                               const float* __restrict__ W_out,
                                               const float* __restrict__ W_bias,
                                               float* __restrict__ bias,
                                               uint16_t* __restrict__ wbf) {
    __shared__ float kern_s[KDIM];
    __shared__ float win_s[KDIM];
    __shared__ float wout_s[COUT];
    const int b = blockIdx.x, tid = threadIdx.x;
    if (tid < KDIM) kern_s[tid] = kern[(size_t)b * KDIM + tid];
    __syncthreads();
    if (tid < KDIM) {  // w_in[b, o, j] = sum_c kern[b,c,j] * W_in[o,c]
        const int o = tid / KW, j = tid - o * KW;
        float s = 0.f;
        for (int c = 0; c < CIN; ++c) s += kern_s[c * KW + j] * W_in[(size_t)o * CIN + c];
        win_s[o * KW + j] = s;
    }
    if (tid < COUT) {  // w_out[b,o], bias[b,o]
        float so = 0.f, sb = 0.f;
        for (int k = 0; k < KDIM; ++k) {
            const float kv = kern_s[k];
            so += kv * W_out[(size_t)tid * KDIM + k];
            sb += kv * W_bias[(size_t)tid * KDIM + k];
        }
        wout_s[tid] = so;
        bias[(size_t)b * COUT + tid] = sb;
    }
    __syncthreads();
    uint16_t* wb = wbf + (size_t)b * COUT * KDIM;
    for (int idx = tid; idx < COUT * KDIM; idx += 256) {
        const int o = idx / KDIM, k = idx - o * KDIM;
        wb[idx] = f2bf(win_s[k] * wout_s[o]);  // weight[b,o,c,t] = w_in[b,c,t]*w_out[b,o]
    }
}

// ---------------------------------------------------------------------------
// Kernel 3: per-sample conv as WMMA GEMM. M=128(COUT), K=192(CIN*3), N=128/block.
// Grid (LEN/128, B), 256 threads = 8 waves; wave w owns M rows [16w,16w+16).
// LDS holds the im2col tile k-major per column so every WMMA B fragment is
// two contiguous ds_load_b128 at immediate offsets; all 8 fragments of a
// k-step are preloaded before the 8 WMMAs so dscnt waits stagger.
// ---------------------------------------------------------------------------
#define XIMST 200  /* row stride in bf16 elems: 192 rounded up, 400B = 16B-aligned */

__global__ __launch_bounds__(256) void k_conv(const float* __restrict__ x,
                                              const uint16_t* __restrict__ wbf,
                                              const float* __restrict__ bias,
                                              float* __restrict__ y) {
    __shared__ __align__(16) uint16_t XimT[128 * XIMST];  // [col][k] 51.2 KB
    __shared__ float bias_s[COUT];

    const int b = blockIdx.y;
    const int l0 = blockIdx.x * 128;
    const int tid = threadIdx.x;
    const int w = tid >> 5;         // wave id (wave32)
    const int lane = tid & 31;
    const int g = lane >> 4;        // half-wave group
    const int lane16 = lane & 15;

    if (tid < COUT) bias_s[tid] = bias[(size_t)b * COUT + tid];

    // Build im2col tile: XimT[col][3c+t] = x[b, c, l0+col+t-1], col in [0,128).
    // Each thread reads one f32 (coalesced), converts once, scatters <=3 u16.
    for (int idx = tid; idx < CIN * 130; idx += 256) {
        const int c = idx / 130, j = idx - c * 130;   // j indexes gcol = l0+j-1
        const int gl = l0 + j - 1;
        const float v = (gl >= 0 && gl < LEN) ? x[((size_t)b * CIN + c) * LEN + gl] : 0.f;
        const uint16_t bv = f2bf(v);
        const int k3 = c * 3;
#pragma unroll
        for (int t = 0; t < KW; ++t) {
            const int col = j - t;                    // col + t - 1 == j - 1
            if (col >= 0 && col < 128) XimT[col * XIMST + k3 + t] = bv;
        }
    }

    // Prefetch next L-tile of x (gfx1250 global_prefetch path).
    if (l0 + 128 < LEN) {
        const int pc = tid >> 2;
        __builtin_prefetch(&x[((size_t)b * CIN + pc) * LEN + l0 + 128 + (tid & 3) * 32], 0, 1);
    }

    // A fragments: wave's 16 rows of the 128x192 bf16 weight matrix, packed
    // contiguous in K (V0:K0,1..V7:K14,15; +16 for g=1) to match B.
    ABfrag afrag[6];
    {
        const uint16_t* wrow = wbf + ((size_t)b * COUT + (w * 16 + lane16)) * KDIM;
#pragma unroll
        for (int kk = 0; kk < 6; ++kk) {
#pragma unroll
            for (int p = 0; p < 8; ++p) {
                const int kbase = 2 * p + 16 * g;
                afrag[kk].u[p] =
                    *reinterpret_cast<const uint32_t*>(wrow + kk * 32 + kbase);
            }
        }
    }

    __syncthreads();

    const v8f vzero = {0.f, 0.f, 0.f, 0.f, 0.f, 0.f, 0.f, 0.f};
    v8f acc[8];
#pragma unroll
    for (int n = 0; n < 8; ++n) acc[n] = vzero;

    // Per-lane LDS base for its column within each N-tile.
    const uint16_t* lrow0 = &XimT[lane16 * XIMST + 16 * g];

#pragma unroll
    for (int kk = 0; kk < 6; ++kk) {
        // Preload all 8 B fragments for this k-step (16x ds_load_b128),
        // then issue the 8 independent WMMAs: waits stagger, latency hides.
        ABfrag barr[8];
#pragma unroll
        for (int n = 0; n < 8; ++n) {
            const uint4* src = reinterpret_cast<const uint4*>(
                lrow0 + n * (16 * XIMST) + kk * 32);
            const uint4 lo = src[0], hi = src[1];
            barr[n].u[0] = lo.x; barr[n].u[1] = lo.y;
            barr[n].u[2] = lo.z; barr[n].u[3] = lo.w;
            barr[n].u[4] = hi.x; barr[n].u[5] = hi.y;
            barr[n].u[6] = hi.z; barr[n].u[7] = hi.w;
        }
#pragma unroll
        for (int n = 0; n < 8; ++n) {
            acc[n] = __builtin_amdgcn_wmma_f32_16x16x32_bf16(
                false, afrag[kk].v, false, barr[n].v, (short)0, acc[n], false, false);
        }
    }

    // D layout: VGPR r -> M = r + 8*g, N = lane16.
#pragma unroll
    for (int n = 0; n < 8; ++n) {
#pragma unroll
        for (int r = 0; r < 8; ++r) {
            const int rowM = w * 16 + r + 8 * g;
            y[((size_t)b * COUT + rowM) * LEN + l0 + n * 16 + lane16] =
                acc[n][r] + bias_s[rowM];
        }
    }
}

// ---------------------------------------------------------------------------
extern "C" void kernel_launch(void* const* d_in, const int* in_sizes, int n_in,
                              void* d_out, int out_size, void* d_ws, size_t ws_size,
                              hipStream_t stream) {
    (void)in_sizes; (void)n_in; (void)out_size; (void)ws_size;
    const float* x      = (const float*)d_in[0];
    const float* Wk     = (const float*)d_in[1];
    const float* W_in   = (const float*)d_in[2];
    const float* W_out  = (const float*)d_in[3];
    const float* W_bias = (const float*)d_in[4];
    float* y = (float*)d_out;

    // Workspace layout (needs ~1.62 MB):
    float*    kern_ws = (float*)d_ws;                       // 32*192 f32
    float*    bias_ws = kern_ws + B_ * KDIM;                // 32*128 f32
    uint16_t* wbf     = (uint16_t*)(bias_ws + B_ * COUT);   // 32*128*192 bf16

    k_rowsum<<<dim3(CIN, B_), 256, 0, stream>>>(x, Wk, kern_ws);
    k_makew<<<B_, 256, 0, stream>>>(kern_ws, W_in, W_out, W_bias, bias_ws, wbf);
    k_conv<<<dim3(LEN / 128, B_), 256, 0, stream>>>(x, wbf, bias_ws, y);
}
